// GraphSage_83038897701149
// MI455X (gfx1250) — compile-verified, matrix operand order checked by reading, not run on hardware
//
#include <hip/hip_runtime.h>

typedef __attribute__((ext_vector_type(16))) _Float16 v16h;
typedef __attribute__((ext_vector_type(8)))  _Float16 v8h;
typedef __attribute__((ext_vector_type(8)))  float    v8f;

#define DIM   128
#define K2    256      // concatenated K = 2*DIM
#define ASTR  264      // LDS row stride in halves (conflict-free, 16B aligned: 264*2=528=33*16)

// ---------------- zero fill (float4 per thread) ----------------
__global__ void gs_zero_f32(float* __restrict__ p, int n4) {
    int i = blockIdx.x * blockDim.x + threadIdx.x;
    if (i < n4) {
        float4 z = {0.f, 0.f, 0.f, 0.f};
        *(float4*)(p + 4 * i) = z;
    }
}

// ---------------- degree: deg[dst[e]] += 1 ----------------
__global__ void gs_degree(const int* __restrict__ dst, float* __restrict__ deg, int E) {
    int e = blockIdx.x * blockDim.x + threadIdx.x;
    if (e < E)
        unsafeAtomicAdd(&deg[dst[e]], 1.0f);   // guaranteed global_atomic_add_f32
}

// ---------------- scatter: acc[dst] += x[src] (4 floats / thread) ----------------
__global__ void gs_scatter(const float* __restrict__ x, const int* __restrict__ src,
                           const int* __restrict__ dst, float* __restrict__ acc, int E) {
    int t = blockIdx.x * blockDim.x + threadIdx.x;
    int e = t >> 5;
    if (e >= E) return;
    int c = (t & 31) * 4;
    const float4 v = *(const float4*)(x + (long)src[e] * DIM + c);
    float* a = acc + (long)dst[e] * DIM + c;
    unsafeAtomicAdd(a + 0, v.x);
    unsafeAtomicAdd(a + 1, v.y);
    unsafeAtomicAdd(a + 2, v.z);
    unsafeAtomicAdd(a + 3, v.w);
}

// ---------------- weight pre-convert: wct[n*256 + k] = f16( [Wself;Wneigh][k][n] ) ----------------
__global__ void gs_convert_w(const float* __restrict__ wself, const float* __restrict__ wneigh,
                             _Float16* __restrict__ wct) {
    int t = blockIdx.x * blockDim.x + threadIdx.x;   // 256*128 threads
    int n = t & (DIM - 1);
    int k = t >> 7;                                   // 0..255
    float v = (k < DIM) ? wself[k * DIM + n] : wneigh[(k - DIM) * DIM + n];
    wct[n * K2 + k] = (_Float16)v;
}

// ---------------- fused SAGE layer GEMM ----------------
// out[m][n] = relu( [h | acc/deg][m][:] @ Wct[:][n] + b[n] ) + h[m][n]
// block = 256 threads (8 waves), 16 rows per block; wave w -> cols [16w, 16w+16)
__launch_bounds__(256, 2)
__global__ void gs_gemm(const float* __restrict__ hin, const float* __restrict__ acc,
                        const float* __restrict__ deg, const _Float16* __restrict__ wct,
                        const float* __restrict__ bias, float* __restrict__ out) {
    __shared__ __align__(16) _Float16 Alds[16 * ASTR];

    const int row0 = blockIdx.x * 16;
    const int tid  = threadIdx.x;

    // Stage A-panel: 16 rows x 256 K (h ++ h_neigh) as f16
    for (int e = tid; e < 16 * K2; e += 256) {
        int r = e >> 8;          // 0..15
        int k = e & (K2 - 1);    // 0..255
        int gr = row0 + r;
        float v;
        if (k < DIM) {
            v = hin[(long)gr * DIM + k];
        } else {
            float d = deg[gr];
            float s = acc[(long)gr * DIM + (k - DIM)];
            v = (d > 0.f) ? (s / d) : 0.f;
        }
        Alds[r * ASTR + k] = (_Float16)v;
    }
    __syncthreads();

    const int lane = tid & 31;
    const int wave = tid >> 5;            // 0..7
    const int n0   = wave * 16;
    const int rr   = lane & 15;           // A row / B column (mod 16)
    const int kb   = (lane >> 4) * 8;     // lanes 16-31 start at K+8
    const int n    = n0 + rr;

    const _Float16* __restrict__ arow = &Alds[rr * ASTR + kb];
    const _Float16* __restrict__ brow = wct + (long)n * K2 + kb;

    v8f c = {};
#pragma unroll
    for (int ks = 0; ks < 8; ++ks) {
        v8h alo = *(const v8h*)(arow + ks * 32);
        v8h ahi = *(const v8h*)(arow + ks * 32 + 16);
        v8h blo = *(const v8h*)(brow + ks * 32);
        v8h bhi = *(const v8h*)(brow + ks * 32 + 16);
        v16h a = __builtin_shufflevector(alo, ahi, 0,1,2,3,4,5,6,7,8,9,10,11,12,13,14,15);
        v16h b = __builtin_shufflevector(blo, bhi, 0,1,2,3,4,5,6,7,8,9,10,11,12,13,14,15);
        c = __builtin_amdgcn_wmma_f32_16x16x32_f16(false, a, false, b, (short)0, c, false, false);
    }

    // Epilogue: bias + relu + residual (residual == layer input)
    const float bn = bias[n];
    const int mbase = row0 + ((lane >> 4) * 8);
#pragma unroll
    for (int j = 0; j < 8; ++j) {
        int m = mbase + j;
        float v = c[j] + bn;
        v = fmaxf(v, 0.f);
        out[(long)m * DIM + n] = v + hin[(long)m * DIM + n];
    }
}

// ---------------- host-side launch ----------------
extern "C" void kernel_launch(void* const* d_in, const int* in_sizes, int n_in,
                              void* d_out, int out_size, void* d_ws, size_t ws_size,
                              hipStream_t stream) {
    const float* feat = (const float*)d_in[0];
    const int*   src  = (const int*)d_in[1];
    const int*   dst  = (const int*)d_in[2];
    // d_in[3] = etype (unused by reference math)
    const float* wself[3]  = {(const float*)d_in[4],  (const float*)d_in[7],  (const float*)d_in[10]};
    const float* wneigh[3] = {(const float*)d_in[5],  (const float*)d_in[8],  (const float*)d_in[11]};
    const float* bias[3]   = {(const float*)d_in[6],  (const float*)d_in[9],  (const float*)d_in[12]};

    const int E = in_sizes[1];
    const int N = in_sizes[0] / DIM;   // 50000

    // Workspace carve-up (256B aligned slots)
    char* ws = (char*)d_ws;
    auto carve = [&](size_t bytes) {
        void* p = (void*)ws;
        ws += (bytes + 255) & ~(size_t)255;
        return p;
    };
    float*     deg = (float*)    carve((size_t)N * 4);
    float*     acc = (float*)    carve((size_t)N * DIM * 4);
    float*     h1  = (float*)    carve((size_t)N * DIM * 4);
    float*     h2  = (float*)    carve((size_t)N * DIM * 4);
    _Float16*  wct = (_Float16*) carve((size_t)3 * K2 * DIM * 2);

    const int nThreads = 256;

    // Degree (computed once, reused by all layers)
    gs_zero_f32<<<(N / 4 + nThreads - 1) / nThreads, nThreads, 0, stream>>>(deg, N / 4);
    gs_degree<<<(E + nThreads - 1) / nThreads, nThreads, 0, stream>>>(dst, deg, E);

    // Pre-convert weights to f16, k-contiguous per output column
    for (int l = 0; l < 3; ++l)
        gs_convert_w<<<(K2 * DIM) / nThreads, nThreads, 0, stream>>>(
            wself[l], wneigh[l], wct + (size_t)l * K2 * DIM);

    const float* layer_in[3]  = {feat, h1, h2};
    float*       layer_out[3] = {h1, h2, (float*)d_out};

    const int accN4    = (N * DIM) / 4;
    const int zBlocks  = (accN4 + nThreads - 1) / nThreads;
    const int sBlocks  = ((E * 32) + nThreads - 1) / nThreads;
    const int gBlocks  = N / 16;   // 50000/16 = 3125, exact

    for (int l = 0; l < 3; ++l) {
        gs_zero_f32<<<zBlocks, nThreads, 0, stream>>>(acc, accN4);
        gs_scatter<<<sBlocks, nThreads, 0, stream>>>(layer_in[l], src, dst, acc, E);
        gs_gemm<<<gBlocks, nThreads, 0, stream>>>(
            layer_in[l], acc, deg, wct + (size_t)l * K2 * DIM, bias[l], layer_out[l]);
    }
}
